// MedTok_13013750907147
// MI455X (gfx1250) — matrix operand order repaired and checked by reading, not compiled, not run on hardware
//
#include <hip/hip_runtime.h>
#include <hip/hip_bf16.h>
#include <math.h>

#define BB     64
#define LL     77
#define WW     768
#define HH     12
#define DH     64
#define NLAY   12
#define NNODE  65536
#define INDIM  128
#define HID    512
#define OUTD   768
#define DPROJ  768
#define NCB    8000
#define SPLITR 4000
#define KTOP   5

typedef long long ll;
typedef __attribute__((ext_vector_type(16))) __bf16 v16bf;
typedef __attribute__((ext_vector_type(8)))  float  v8f;

__device__ __forceinline__ unsigned short f2bf(float f) {
  union { float f; unsigned u; } v; v.f = f;
  unsigned r = v.u + 0x7FFFu + ((v.u >> 16) & 1u);  // round-to-nearest-even
  return (unsigned short)(r >> 16);
}

union Frag {
  v16bf v;
  unsigned short u[16];
  uint4 q[2];
};

// gfx1250 async global->LDS copy (ISA 15.18.3 op 98), ASYNCcnt-tracked.
// vdst = 32-bit LDS byte address (low word of generic pointer), vaddr = 64-bit
// global address. Followed later by s_wait_asynccnt before the barrier.
__device__ __forceinline__ void async_ld_b128(const unsigned short* g, unsigned lds_addr) {
#if defined(__HIP_DEVICE_COMPILE__)
  asm volatile("global_load_async_to_lds_b128 %0, %1, off"
               :: "v"(lds_addr), "v"(g) : "memory");
#endif
}
__device__ __forceinline__ void wait_asynccnt0() {
#if defined(__HIP_DEVICE_COMPILE__)
  asm volatile("s_wait_asynccnt 0x0" ::: "memory");
#endif
}

// ---------------------------------------------------------------------------
// Generic batched WMMA GEMM:  C = act(A(bf16)[M,K] * B(bf16) + bias) + resid
// B is [K,N] (transB=0) or [N,K] (transB=1). Block tile 64(M) x 64(N);
// 8 waves in 2x4; each wave computes 32x16 of C (two 16x16 WMMA tiles sharing
// one B fragment). B tile staged into LDS as [n][k] (padded stride 40 ->
// conflict-free ds_load_b128). transB fast path uses global_load_async_to_lds
// (one b128 per thread, no VGPR round trip); transB=0 or edge tiles use the
// VGPR/scalar path. A fragments are two global b128 loads on the interior
// fast path (+ global_prefetch of the next K block).
// ---------------------------------------------------------------------------
#define PB 40   // padded LDS row stride in bf16 elements (80B = 16B aligned)

__global__ void __launch_bounds__(256)
wmma_gemm(const unsigned short* __restrict__ A, ll lda, ll sA1, ll sA2,
          const unsigned short* __restrict__ B, ll ldb, ll sB1, ll sB2, int transB,
          const float* __restrict__ bias,
          const float* __restrict__ resid, ll ldr,
          float* __restrict__ Cf, unsigned short* __restrict__ Cb,
          ll ldc, ll sC1, ll sC2,
          int M, int N, int K, int b2, int act)
{
  __shared__ unsigned short ldsB[64 * PB];

  int lane = threadIdx.x & 31;
  int wv   = threadIdx.x >> 5;         // 0..7
  int wm   = wv >> 2, wn = wv & 3;     // 2 (M) x 4 (N) waves
  int tmBase = blockIdx.y * 64 + wm * 32;   // wave covers rows [tmBase, tmBase+32)
  int tn0 = blockIdx.x * 64;                // block cols
  int tn  = tn0 + wn * 16;                  // wave cols

  int z  = blockIdx.z;
  int i1 = z / b2, i2 = z - i1 * b2;
  const unsigned short* Ab = A + i1 * sA1 + i2 * sA2;
  const unsigned short* Bb = B + i1 * sB1 + i2 * sB2;

  int hl = lane >> 4;                  // half: lanes 0-15 vs 16-31
  int r  = lane & 15;
  int cnL = wn * 16 + r;               // local column in block tile (0..63)
  v8f acc0 = {}, acc1 = {};

  for (int k0 = 0; k0 < K; k0 += 32) {
    __syncthreads();
    // ---- stage B tile (32K x 64N) into LDS as ldsB[n][k] ----
    if (transB) {
      int n  = threadIdx.x >> 2;           // 0..63
      int kc = (threadIdx.x & 3) * 8;      // 0,8,16,24
      int gn = tn0 + n;
      unsigned laddr = (unsigned)(unsigned long long)(&ldsB[n * PB + kc]);
      if (((ldb & 7) == 0) && (gn < N) && (k0 + kc + 8 <= K)) {
        async_ld_b128(Bb + (ll)gn * ldb + k0 + kc, laddr);   // async DMA to LDS
      } else {
#pragma unroll
        for (int j = 0; j < 8; ++j) {
          int gk = k0 + kc + j;
          ldsB[n * PB + kc + j] =
              (gn < N && gk < K) ? Bb[(ll)gn * ldb + gk] : (unsigned short)0;
        }
      }
    } else {
      int kk = threadIdx.x >> 3;           // 0..31
      int nc = (threadIdx.x & 7) * 8;      // 0,8,...,56
      int gk = k0 + kk;
      Frag t;
      if (((ldb & 7) == 0) && (gk < K) && (tn0 + nc + 8 <= N)) {
        t.q[0] = *(const uint4*)(Bb + (ll)gk * ldb + tn0 + nc);
      } else {
#pragma unroll
        for (int j = 0; j < 8; ++j) {
          int gn = tn0 + nc + j;
          t.u[j] = (gk < K && gn < N) ? Bb[(ll)gk * ldb + gn] : (unsigned short)0;
        }
      }
#pragma unroll
      for (int j = 0; j < 8; ++j) ldsB[(nc + j) * PB + kk] = t.u[j];
    }
    wait_asynccnt0();
    __syncthreads();

    // ---- B fragment from LDS: lane = column, K = half*16 + j ----
    Frag bfr;
    bfr.q[0] = *(const uint4*)&ldsB[cnL * PB + hl * 16];
    bfr.q[1] = *(const uint4*)&ldsB[cnL * PB + hl * 16 + 8];

    // ---- A fragments (rows tmBase+r and tmBase+16+r) ----
    Frag a0, a1;
    bool fastA = ((lda & 7) == 0) && (tmBase + 32 <= M) && (k0 + 32 <= K);
    if (fastA) {
      const unsigned short* p0 = Ab + (ll)(tmBase + r) * lda + k0 + hl * 16;
      const unsigned short* p1 = Ab + (ll)(tmBase + 16 + r) * lda + k0 + hl * 16;
      a0.q[0] = *(const uint4*)p0;
      a0.q[1] = *(const uint4*)(p0 + 8);
      a1.q[0] = *(const uint4*)p1;
      a1.q[1] = *(const uint4*)(p1 + 8);
      if (k0 + 64 <= K) {                       // gfx1250 global_prefetch_b8
        __builtin_prefetch(p0 + 32, 0, 1);
        __builtin_prefetch(p1 + 32, 0, 1);
      }
    } else {
      int row0 = tmBase + r, row1 = tmBase + 16 + r;
#pragma unroll
      for (int j = 0; j < 16; ++j) {
        int ka = k0 + j + hl * 16;   // A layout: K = j + half*16
        bool kin = ka < K;
        a0.u[j] = (row0 < M && kin) ? Ab[(ll)row0 * lda + ka] : (unsigned short)0;
        a1.u[j] = (row1 < M && kin) ? Ab[(ll)row1 * lda + ka] : (unsigned short)0;
      }
    }

    acc0 = __builtin_amdgcn_wmma_f32_16x16x32_bf16(false, a0.v, false, bfr.v,
                                                   (short)0, acc0, false, false);
    acc1 = __builtin_amdgcn_wmma_f32_16x16x32_bf16(false, a1.v, false, bfr.v,
                                                   (short)0, acc1, false, false);
  }

  // ---- epilogue: two 16x16 tiles ----
  float* Cfp = Cf ? Cf + i1 * sC1 + i2 * sC2 : (float*)0;
  unsigned short* Cbp = Cb ? Cb + i1 * sC1 + i2 * sC2 : (unsigned short*)0;
  const float* Rp = resid;
  int col = tn + r;
#pragma unroll
  for (int t = 0; t < 2; ++t) {
    v8f acc = t ? acc1 : acc0;
    int tb = tmBase + t * 16;
#pragma unroll
    for (int v = 0; v < 8; ++v) {
      int rowc = tb + v + hl * 8;        // C/D: VGPR v -> row v (+8 upper half)
      if (rowc < M && col < N) {
        float val = acc[v];
        if (bias) val += bias[col];
        if (act == 1) val = val / (1.0f + __expf(-1.702f * val)); // quick-gelu
        else if (act == 2) val = fmaxf(val, 0.0f);                // relu
        if (Rp) val += Rp[(ll)rowc * ldr + col];
        if (Cfp) Cfp[(ll)rowc * ldc + col] = val;
        if (Cbp) Cbp[(ll)rowc * ldc + col] = f2bf(val);
      }
    }
  }
}

// ------------------------- elementwise / reduction kernels ------------------
__global__ void cvt_bf_k(const float* __restrict__ in, unsigned short* __restrict__ out, ll n) {
  ll i = (ll)blockIdx.x * blockDim.x + threadIdx.x;
  ll st = (ll)gridDim.x * blockDim.x;
  for (; i < n; i += st) out[i] = f2bf(in[i]);
}

// batched transpose-convert: in = [Lb][Kd][Nd] f32 -> out = [Lb][Nd][Kd] bf16
__global__ void cvt_tr_k(const float* __restrict__ in, unsigned short* __restrict__ out,
                         int Kd, int Nd, ll total) {
  ll i = (ll)blockIdx.x * blockDim.x + threadIdx.x;
  ll st = (ll)gridDim.x * blockDim.x;
  ll mat = (ll)Kd * Nd;
  for (; i < total; i += st) {
    ll l = i / mat; ll rem = i - l * mat;
    int k = (int)(rem / Nd), n = (int)(rem % Nd);
    out[l * mat + (ll)n * Kd + k] = f2bf(in[i]);
  }
}

__global__ void fill_k(float* p, float v, ll n) {
  ll i = (ll)blockIdx.x * blockDim.x + threadIdx.x;
  ll st = (ll)gridDim.x * blockDim.x;
  for (; i < n; i += st) p[i] = v;
}

__global__ void embed_k(const int* __restrict__ text, const float* __restrict__ tok,
                        const float* __restrict__ pos, float* __restrict__ x) {
  ll i = (ll)blockIdx.x * blockDim.x + threadIdx.x;
  if (i >= (ll)BB * LL * WW) return;
  int w = (int)(i % WW);
  ll bl = i / WW;
  int l = (int)(bl % LL);
  x[i] = tok[(ll)text[bl] * WW + w] + pos[(ll)l * WW + w];
}

__global__ void ln_k(const float* __restrict__ x, const float* __restrict__ g,
                     const float* __restrict__ b, float* __restrict__ outf,
                     unsigned short* __restrict__ outb, int D) {
  int row = blockIdx.x;
  const float* xr = x + (ll)row * D;
  __shared__ float sh[256];
  float s = 0.f;
  for (int i = threadIdx.x; i < D; i += blockDim.x) s += xr[i];
  sh[threadIdx.x] = s; __syncthreads();
  for (int o = 128; o > 0; o >>= 1) { if (threadIdx.x < o) sh[threadIdx.x] += sh[threadIdx.x + o]; __syncthreads(); }
  float mean = sh[0] / (float)D; __syncthreads();
  float v = 0.f;
  for (int i = threadIdx.x; i < D; i += blockDim.x) { float d = xr[i] - mean; v += d * d; }
  sh[threadIdx.x] = v; __syncthreads();
  for (int o = 128; o > 0; o >>= 1) { if (threadIdx.x < o) sh[threadIdx.x] += sh[threadIdx.x + o]; __syncthreads(); }
  float rstd = rsqrtf(sh[0] / (float)D + 1e-5f);
  for (int i = threadIdx.x; i < D; i += blockDim.x) {
    float val = (xr[i] - mean) * rstd * g[i] + b[i];
    if (outf) outf[(ll)row * D + i] = val;
    if (outb) outb[(ll)row * D + i] = f2bf(val);
  }
}

// row softmax (optionally causal with period Lq), scaled; bf16 probs out
__global__ void softmax_k(const float* __restrict__ S, unsigned short* __restrict__ P,
                          int cols, int ld, float scale, int causal, int Lq) {
  int row = blockIdx.x;
  int lim = causal ? ((row % Lq) + 1) : cols;
  const float* sr = S + (ll)row * ld;
  __shared__ float sh[128];
  float m = -1e30f;
  for (int i = threadIdx.x; i < lim; i += blockDim.x) m = fmaxf(m, sr[i] * scale);
  sh[threadIdx.x] = m; __syncthreads();
  for (int o = 64; o > 0; o >>= 1) { if (threadIdx.x < o) sh[threadIdx.x] = fmaxf(sh[threadIdx.x], sh[threadIdx.x + o]); __syncthreads(); }
  m = sh[0]; __syncthreads();
  float s = 0.f;
  for (int i = threadIdx.x; i < lim; i += blockDim.x) s += __expf(sr[i] * scale - m);
  sh[threadIdx.x] = s; __syncthreads();
  for (int o = 64; o > 0; o >>= 1) { if (threadIdx.x < o) sh[threadIdx.x] += sh[threadIdx.x + o]; __syncthreads(); }
  float inv = 1.0f / sh[0];
  for (int i = threadIdx.x; i < cols; i += blockDim.x) {
    float p = (i < lim) ? __expf(sr[i] * scale - m) * inv : 0.0f;
    P[(ll)row * ld + i] = f2bf(p);
  }
}

__global__ void eos_k(const int* __restrict__ text, int* __restrict__ eos) {
  int b = threadIdx.x;
  if (b >= BB) return;
  int best = -2147483647 - 1, bi = 0;
  for (int l = 0; l < LL; ++l) { int t = text[b * LL + l]; if (t > best) { best = t; bi = l; } }
  eos[b] = bi;
}

__global__ void gather_k(const unsigned short* __restrict__ xb, const int* __restrict__ eos,
                         unsigned short* __restrict__ out) {
  int i = blockIdx.x * blockDim.x + threadIdx.x;
  if (i >= BB * WW) return;
  int b = i / WW, w = i % WW;
  out[i] = xb[((ll)(b * LL + eos[b])) * WW + w];
}

__global__ void deg_k(const int* __restrict__ dst, float* __restrict__ deg, ll E) {
  ll i = (ll)blockIdx.x * blockDim.x + threadIdx.x;
  ll st = (ll)gridDim.x * blockDim.x;
  for (; i < E; i += st) atomicAdd(&deg[dst[i]], 1.0f);
}

__global__ void rsqrt_k(const float* __restrict__ deg, float* __restrict__ dis, int n) {
  int i = blockIdx.x * blockDim.x + threadIdx.x;
  if (i < n) dis[i] = rsqrtf(deg[i]);
}

__global__ void agg_self_k(const float* __restrict__ hw, const float* __restrict__ dis,
                           float* __restrict__ agg, int D, ll n) {
  ll i = (ll)blockIdx.x * blockDim.x + threadIdx.x;
  ll st = (ll)gridDim.x * blockDim.x;
  for (; i < n; i += st) {
    int node = (int)(i / D);
    float ds = dis[node];
    agg[i] = hw[i] * ds * ds;
  }
}

__global__ void scatter_k(const int* __restrict__ src, const int* __restrict__ dst,
                          const float* __restrict__ hw, const float* __restrict__ dis,
                          float* __restrict__ agg, int D, ll total) {
  ll i = (ll)blockIdx.x * blockDim.x + threadIdx.x;
  ll st = (ll)gridDim.x * blockDim.x;
  for (; i < total; i += st) {
    ll e = i / D; int d = (int)(i % D);
    int s = src[e], dd = dst[e];
    atomicAdd(&agg[(ll)dd * D + d], hw[(ll)s * D + d] * dis[s] * dis[dd]);
  }
}

__global__ void relu_bf_k(const float* __restrict__ in, unsigned short* __restrict__ out, ll n) {
  ll i = (ll)blockIdx.x * blockDim.x + threadIdx.x;
  ll st = (ll)gridDim.x * blockDim.x;
  for (; i < n; i += st) out[i] = f2bf(fmaxf(in[i], 0.0f));
}

__global__ void cnt_k(const int* __restrict__ nb, float* __restrict__ cnt, int n) {
  int i = blockIdx.x * blockDim.x + threadIdx.x;
  if (i < n) atomicAdd(&cnt[nb[i]], 1.0f);
}

__global__ void segsum_k(const int* __restrict__ nb, const float* __restrict__ g,
                         float* __restrict__ sums, int D, ll n) {
  ll i = (ll)blockIdx.x * blockDim.x + threadIdx.x;
  ll st = (ll)gridDim.x * blockDim.x;
  for (; i < n; i += st) {
    int node = (int)(i / D); int d = (int)(i % D);
    atomicAdd(&sums[(ll)nb[node] * D + d], g[i]);
  }
}

__global__ void mean_k(const float* __restrict__ sums, const float* __restrict__ cnt,
                       float* __restrict__ out, int D, int n) {
  int i = blockIdx.x * blockDim.x + threadIdx.x;
  if (i >= n) return;
  out[i] = sums[i] / fmaxf(cnt[i / D], 1.0f);
}

__global__ void sumsq_k(const float* __restrict__ A, float* __restrict__ out, int D) {
  int row = blockIdx.x;
  const float* a = A + (ll)row * D;
  __shared__ float sh[256];
  float s = 0.f;
  for (int i = threadIdx.x; i < D; i += blockDim.x) s += a[i] * a[i];
  sh[threadIdx.x] = s; __syncthreads();
  for (int o = 128; o > 0; o >>= 1) { if (threadIdx.x < o) sh[threadIdx.x] += sh[threadIdx.x + o]; __syncthreads(); }
  if (threadIdx.x == 0) out[row] = sh[0];
}

// soft-VQ: per row, d = sqrt(max(|e|^2+|c|^2-2G,1e-12)); top-5 smallest (first-index
// tiebreak), softmax(-d), weighted codebook sum; idx emitted as float
__global__ void vq_k(const float* __restrict__ G, int ldg, const float* __restrict__ Ef,
                     const float* __restrict__ cn, const float* __restrict__ cbA,
                     const float* __restrict__ cbB, int splitRows,
                     float* __restrict__ qout, float* __restrict__ idxout,
                     int rows, int D, int Ncb) {
  int b = blockIdx.x * blockDim.x + threadIdx.x;
  if (b >= rows) return;
  const float* er = Ef + (ll)b * D;
  float en = 0.f;
  for (int i = 0; i < D; ++i) en += er[i] * er[i];
  float bd[KTOP]; int bi[KTOP];
  for (int k = 0; k < KTOP; ++k) { bd[k] = 1e30f; bi[k] = 0; }
  const float* gr = G + (ll)b * ldg;
  for (int j = 0; j < Ncb; ++j) {
    float d2 = en + cn[j] - 2.0f * gr[j];
    float d = sqrtf(fmaxf(d2, 1e-12f));
    if (d < bd[KTOP - 1]) {
      int p = KTOP - 1;
      while (p > 0 && d < bd[p - 1]) { bd[p] = bd[p - 1]; bi[p] = bi[p - 1]; --p; }
      bd[p] = d; bi[p] = j;
    }
  }
  float m = -bd[0], w[KTOP], s = 0.f;
  for (int k = 0; k < KTOP; ++k) { w[k] = __expf(-bd[k] - m); s += w[k]; }
  for (int k = 0; k < KTOP; ++k) w[k] /= s;
  for (int d = 0; d < D; ++d) {
    float acc = 0.f;
    for (int k = 0; k < KTOP; ++k) {
      int j = bi[k];
      const float* cr = (j < splitRows) ? (cbA + (ll)j * D) : (cbB + (ll)(j - splitRows) * D);
      acc += w[k] * cr[d];
    }
    qout[(ll)b * D + d] = acc;
  }
  for (int k = 0; k < KTOP; ++k) idxout[b * KTOP + k] = (float)bi[k];
}

// ------------------------------ host helpers --------------------------------
static void gemm(hipStream_t s,
                 const unsigned short* A, ll lda, ll sA1, ll sA2,
                 const unsigned short* B, ll ldb, ll sB1, ll sB2, int tB,
                 const float* bias, const float* resid, ll ldr,
                 float* Cf, unsigned short* Cb, ll ldc, ll sC1, ll sC2,
                 int M, int N, int K, int b1, int b2, int act) {
  dim3 g((unsigned)((N + 63) / 64), (unsigned)((M + 63) / 64), (unsigned)(b1 * b2));
  wmma_gemm<<<g, 256, 0, s>>>(A, lda, sA1, sA2, B, ldb, sB1, sB2, tB, bias, resid, ldr,
                              Cf, Cb, ldc, sC1, sC2, M, N, K, b2, act);
}

static void cvt(hipStream_t s, const float* in, unsigned short* out, ll n) {
  ll b = (n + 255) / 256; if (b > 16384) b = 16384;
  cvt_bf_k<<<(unsigned)b, 256, 0, s>>>(in, out, n);
}

// transpose-convert: in = [Lb][Kd][Nd] f32 -> out = [Lb][Nd][Kd] bf16
static void cvtT(hipStream_t s, const float* in, unsigned short* out, int Kd, int Nd, int Lb) {
  ll n = (ll)Lb * Kd * Nd;
  ll b = (n + 255) / 256; if (b > 16384) b = 16384;
  cvt_tr_k<<<(unsigned)b, 256, 0, s>>>(in, out, Kd, Nd, n);
}

extern "C" void kernel_launch(void* const* d_in, const int* in_sizes, int n_in,
                              void* d_out, int out_size, void* d_ws, size_t ws_size,
                              hipStream_t stream) {
  (void)n_in; (void)out_size; (void)ws_size;
  const int*   text       = (const int*)d_in[0];
  const int*   edge_index = (const int*)d_in[1];
  const int*   node_batch = (const int*)d_in[2];
  const float* node_feats = (const float*)d_in[3];
  const float* tok_emb    = (const float*)d_in[4];
  const float* pos_emb    = (const float*)d_in[5];
  const float* ln1_g = (const float*)d_in[6];  const float* ln1_b = (const float*)d_in[7];
  const float* attn_in_w  = (const float*)d_in[8];  const float* attn_in_b  = (const float*)d_in[9];
  const float* attn_out_w = (const float*)d_in[10]; const float* attn_out_b = (const float*)d_in[11];
  const float* ln2_g = (const float*)d_in[12]; const float* ln2_b = (const float*)d_in[13];
  const float* fc_w  = (const float*)d_in[14]; const float* fc_b  = (const float*)d_in[15];
  const float* prj_w = (const float*)d_in[16]; const float* prj_b = (const float*)d_in[17];
  const float* lnf_g = (const float*)d_in[18]; const float* lnf_b = (const float*)d_in[19];
  const float* text_proj = (const float*)d_in[20];
  const float* gcn_w0 = (const float*)d_in[21]; const float* gcn_b0 = (const float*)d_in[22];
  const float* gcn_w1 = (const float*)d_in[23]; const float* gcn_b1 = (const float*)d_in[24];
  const float* gcn_w2 = (const float*)d_in[25]; const float* gcn_b2 = (const float*)d_in[26];
  const float* glnf_g = (const float*)d_in[27]; const float* glnf_b = (const float*)d_in[28];
  const float* txtp_w = (const float*)d_in[29]; const float* txtp_b = (const float*)d_in[30];
  const float* grp_w  = (const float*)d_in[31]; const float* grp_b  = (const float*)d_in[32];
  const float* wq_w = (const float*)d_in[33]; const float* wq_b = (const float*)d_in[34];
  const float* wk_w = (const float*)d_in[35]; const float* wk_b = (const float*)d_in[36];
  const float* wv_w = (const float*)d_in[37]; const float* wv_b = (const float*)d_in[38];
  const float* cb_text   = (const float*)d_in[39];
  const float* cb_graph  = (const float*)d_in[40];
  const float* cb_shared = (const float*)d_in[41];
  const ll E = (ll)in_sizes[1] / 2;
  const int* e_src = edge_index;
  const int* e_dst = edge_index + E;
  float* out = (float*)d_out;

  // ---- bump allocator over workspace ----
  char* wsb = (char*)d_ws;
  size_t off = 0;
  auto allocB = [&](size_t bytes) -> void* {
    void* p = wsb + off; off += (bytes + 255) & ~(size_t)255; return p;
  };
  auto aU = [&](ll n) { return (unsigned short*)allocB((size_t)n * 2); };
  auto aF = [&](ll n) { return (float*)allocB((size_t)n * 4); };
  auto aI = [&](ll n) { return (int*)allocB((size_t)n * 4); };

  // bf16 weights, stored TRANSPOSED ([N][K]) so GEMMs use the async-LDS path
  unsigned short* wb_ain = aU((ll)NLAY * WW * 3 * WW);   // [l][3W][W]
  unsigned short* wb_aout= aU((ll)NLAY * WW * WW);       // [l][W][W]
  unsigned short* wb_fc  = aU((ll)NLAY * WW * 4 * WW);   // [l][4W][W]
  unsigned short* wb_prj = aU((ll)NLAY * 4 * WW * WW);   // [l][W][4W]
  unsigned short* wb_tp  = aU((ll)WW * WW);
  unsigned short* wb_txtp= aU((ll)WW * DPROJ);
  unsigned short* wb_grp = aU((ll)OUTD * DPROJ);
  unsigned short* wb_wq  = aU((ll)DPROJ * DPROJ);
  unsigned short* wb_wk  = aU((ll)DPROJ * DPROJ);
  unsigned short* wb_wv  = aU((ll)DPROJ * DPROJ);
  unsigned short* wb_g0  = aU((ll)INDIM * HID);          // [HID][INDIM]
  unsigned short* wb_g1  = aU((ll)HID * HID);
  unsigned short* wb_g2  = aU((ll)HID * OUTD);           // [OUTD][HID]
  unsigned short* cbt_bf = aU((ll)NCB * DPROJ);          // already [N][K]
  unsigned short* cbg_bf = aU((ll)NCB * DPROJ);
  float* cn_t = aF(NCB);
  float* cn_g = aF(NCB);
  // text tower activations
  const ll NT = (ll)BB * LL;            // 4928 token rows
  float*          x_f     = aF(NT * WW);
  unsigned short* h_bf    = aU(NT * WW);
  unsigned short* qkv_bf  = aU(NT * 3 * WW);
  float*          scores  = aF((ll)BB * HH * LL * LL);
  unsigned short* probs_bf= aU((ll)BB * HH * LL * LL);
  unsigned short* o_bf    = aU(NT * WW);
  unsigned short* mlp_bf  = aU(NT * 4 * WW);
  unsigned short* xln_bf  = aU(NT * WW);
  int*            eos     = aI(BB);
  unsigned short* te0_bf  = aU((ll)BB * WW);
  float*          te1_f   = aF((ll)BB * WW);
  unsigned short* te_bf   = aU((ll)BB * WW);
  // graph
  float*          deg     = aF(NNODE);
  float*          dis     = aF(NNODE);
  unsigned short* nf_bf   = aU((ll)NNODE * INDIM);
  float*          hw_f    = aF((ll)NNODE * OUTD);
  float*          agg_f   = aF((ll)NNODE * OUTD);
  unsigned short* g_bf    = aU((ll)NNODE * OUTD);
  float*          sums    = aF((ll)BB * OUTD);
  float*          cnt     = aF(BB);
  float*          ge_f    = aF((ll)BB * OUTD);
  unsigned short* gbe_bf  = aU((ll)BB * OUTD);
  // projections / cross-attn
  unsigned short* tspec_bf = aU((ll)BB * DPROJ);
  unsigned short* gspec_bf = aU((ll)BB * DPROJ);
  unsigned short* tq_bf = aU((ll)BB * DPROJ);
  unsigned short* tk_bf = aU((ll)BB * DPROJ);
  unsigned short* tv_bf = aU((ll)BB * DPROJ);
  unsigned short* gq_bf = aU((ll)BB * DPROJ);
  unsigned short* gk_bf = aU((ll)BB * DPROJ);
  unsigned short* gv_bf = aU((ll)BB * DPROJ);
  float*          s_f   = aF((ll)BB * BB);
  unsigned short* sp_bf = aU((ll)BB * BB);
  unsigned short* tcross_bf = aU((ll)BB * DPROJ);
  unsigned short* gcross_bf = aU((ll)BB * DPROJ);
  float*          G_f   = aF((ll)BB * NCB);

  // ---- weight conversions to bf16 (transposed to [N][K]) ----
  cvtT(stream, attn_in_w,  wb_ain, WW, 3 * WW, NLAY);
  cvtT(stream, attn_out_w, wb_aout, WW, WW, NLAY);
  cvtT(stream, fc_w,  wb_fc,  WW, 4 * WW, NLAY);
  cvtT(stream, prj_w, wb_prj, 4 * WW, WW, NLAY);
  cvtT(stream, text_proj, wb_tp, WW, WW, 1);
  cvtT(stream, txtp_w, wb_txtp, WW, DPROJ, 1);
  cvtT(stream, grp_w,  wb_grp,  OUTD, DPROJ, 1);
  cvtT(stream, wq_w, wb_wq, DPROJ, DPROJ, 1);
  cvtT(stream, wk_w, wb_wk, DPROJ, DPROJ, 1);
  cvtT(stream, wv_w, wb_wv, DPROJ, DPROJ, 1);
  cvtT(stream, gcn_w0, wb_g0, INDIM, HID, 1);
  cvtT(stream, gcn_w1, wb_g1, HID, HID, 1);
  cvtT(stream, gcn_w2, wb_g2, HID, OUTD, 1);
  cvt(stream, cb_text,   cbt_bf,                     (ll)SPLITR * DPROJ);
  cvt(stream, cb_shared, cbt_bf + (ll)SPLITR * DPROJ,(ll)SPLITR * DPROJ);
  cvt(stream, cb_graph,  cbg_bf,                     (ll)SPLITR * DPROJ);
  cvt(stream, cb_shared, cbg_bf + (ll)SPLITR * DPROJ,(ll)SPLITR * DPROJ);
  cvt(stream, node_feats, nf_bf, (ll)NNODE * INDIM);
  sumsq_k<<<SPLITR, 256, 0, stream>>>(cb_text,   cn_t,          DPROJ);
  sumsq_k<<<SPLITR, 256, 0, stream>>>(cb_shared, cn_t + SPLITR, DPROJ);
  sumsq_k<<<SPLITR, 256, 0, stream>>>(cb_graph,  cn_g,          DPROJ);
  sumsq_k<<<SPLITR, 256, 0, stream>>>(cb_shared, cn_g + SPLITR, DPROJ);

  // ================= CLIP text tower =================
  {
    ll n = NT * WW;
    embed_k<<<(unsigned)((n + 255) / 256), 256, 0, stream>>>(text, tok_emb, pos_emb, x_f);
  }
  const ll sTok = (ll)LL * 3 * WW;  // per-image stride in qkv
  for (int i = 0; i < NLAY; ++i) {
    ln_k<<<(unsigned)NT, 256, 0, stream>>>(x_f, ln1_g + i * WW, ln1_b + i * WW, (float*)0, h_bf, WW);
    // qkv = h @ Win + b   (Win^T stored [3W][W], async-LDS path)
    gemm(stream, h_bf, WW, 0, 0, wb_ain + (ll)i * WW * 3 * WW, WW, 0, 0, 1,
         attn_in_b + i * 3 * WW, (float*)0, 0, (float*)0, qkv_bf, 3 * WW, 0, 0,
         (int)NT, 3 * WW, WW, 1, 1, 0);
    // scores[b,h] = q @ k^T  (batched over b x h via strides)
    gemm(stream, qkv_bf, 3 * WW, sTok, DH,
         qkv_bf + WW, 3 * WW, sTok, DH, 1,
         (float*)0, (float*)0, 0,
         scores, (unsigned short*)0, LL, (ll)HH * LL * LL, (ll)LL * LL,
         LL, LL, DH, BB, HH, 0);
    softmax_k<<<(unsigned)(BB * HH * LL), 128, 0, stream>>>(scores, probs_bf, LL, LL, 0.125f, 1, LL);
    // o[b,h] = probs @ v   -> packed as [B, L, W]
    gemm(stream, probs_bf, LL, (ll)HH * LL * LL, (ll)LL * LL,
         qkv_bf + 2 * WW, 3 * WW, sTok, DH, 0,
         (float*)0, (float*)0, 0,
         (float*)0, o_bf, WW, (ll)LL * WW, DH,
         LL, DH, LL, BB, HH, 0);
    // x += o @ Wout + b
    gemm(stream, o_bf, WW, 0, 0, wb_aout + (ll)i * WW * WW, WW, 0, 0, 1,
         attn_out_b + i * WW, x_f, WW, x_f, (unsigned short*)0, WW, 0, 0,
         (int)NT, WW, WW, 1, 1, 0);
    ln_k<<<(unsigned)NT, 256, 0, stream>>>(x_f, ln2_g + i * WW, ln2_b + i * WW, (float*)0, h_bf, WW);
    // mlp = quickgelu(h @ fc + b)
    gemm(stream, h_bf, WW, 0, 0, wb_fc + (ll)i * WW * 4 * WW, WW, 0, 0, 1,
         fc_b + i * 4 * WW, (float*)0, 0, (float*)0, mlp_bf, 4 * WW, 0, 0,
         (int)NT, 4 * WW, WW, 1, 1, 1);
    // x += mlp @ prj + b
    gemm(stream, mlp_bf, 4 * WW, 0, 0, wb_prj + (ll)i * 4 * WW * WW, 4 * WW, 0, 0, 1,
         prj_b + i * WW, x_f, WW, x_f, (unsigned short*)0, WW, 0, 0,
         (int)NT, WW, 4 * WW, 1, 1, 0);
  }
  ln_k<<<(unsigned)NT, 256, 0, stream>>>(x_f, lnf_g, lnf_b, (float*)0, xln_bf, WW);
  eos_k<<<1, 64, 0, stream>>>(text, eos);
  gather_k<<<(BB * WW + 255) / 256, 256, 0, stream>>>(xln_bf, eos, te0_bf);
  gemm(stream, te0_bf, WW, 0, 0, wb_tp, WW, 0, 0, 1,
       (float*)0, (float*)0, 0, te1_f, (unsigned short*)0, WW, 0, 0, BB, WW, WW, 1, 1, 0);
  ln_k<<<BB, 256, 0, stream>>>(te1_f, lnf_g, lnf_b, (float*)0, te_bf, WW);

  // ================= GCN graph tower =================
  fill_k<<<256, 256, 0, stream>>>(deg, 1.0f, NNODE);
  deg_k<<<16384, 256, 0, stream>>>(e_dst, deg, E);
  rsqrt_k<<<NNODE / 256, 256, 0, stream>>>(deg, dis, NNODE);
  const unsigned short* gw[3] = { wb_g0, wb_g1, wb_g2 };
  const float* gb[3] = { gcn_b0, gcn_b1, gcn_b2 };
  int gdin[3] = { INDIM, HID, HID }, gdout[3] = { HID, HID, OUTD };
  const unsigned short* gin = nf_bf;
  for (int l = 0; l < 3; ++l) {
    int Din = gdin[l], Dout = gdout[l];
    gemm(stream, gin, Din, 0, 0, gw[l], Din, 0, 0, 1,
         gb[l], (float*)0, 0, hw_f, (unsigned short*)0, Dout, 0, 0,
         NNODE, Dout, Din, 1, 1, 0);
    agg_self_k<<<16384, 256, 0, stream>>>(hw_f, dis, agg_f, Dout, (ll)NNODE * Dout);
    scatter_k<<<32768, 256, 0, stream>>>(e_src, e_dst, hw_f, dis, agg_f, Dout, E * Dout);
    if (l < 2) { relu_bf_k<<<16384, 256, 0, stream>>>(agg_f, g_bf, (ll)NNODE * Dout); gin = g_bf; }
  }
  fill_k<<<64, 256, 0, stream>>>(sums, 0.0f, (ll)BB * OUTD);
  fill_k<<<1, 64, 0, stream>>>(cnt, 0.0f, BB);
  cnt_k<<<NNODE / 256, 256, 0, stream>>>(node_batch, cnt, NNODE);
  segsum_k<<<16384, 256, 0, stream>>>(node_batch, agg_f, sums, OUTD, (ll)NNODE * OUTD);
  mean_k<<<(BB * OUTD + 255) / 256, 256, 0, stream>>>(sums, cnt, ge_f, OUTD, BB * OUTD);
  ln_k<<<BB, 256, 0, stream>>>(ge_f, glnf_g, glnf_b, (float*)0, gbe_bf, OUTD);

  // ================= projections + cross-attention =================
  float* o_tspec = out + 0;
  float* o_gspec = out + 1 * BB * DPROJ;
  float* o_tcross= out + 2 * BB * DPROJ;
  float* o_gcross= out + 3 * BB * DPROJ;
  gemm(stream, te_bf, WW, 0, 0, wb_txtp, WW, 0, 0, 1,
       txtp_b, (float*)0, 0, o_tspec, tspec_bf, DPROJ, 0, 0, BB, DPROJ, WW, 1, 1, 0);
  gemm(stream, gbe_bf, OUTD, 0, 0, wb_grp, OUTD, 0, 0, 1,
       grp_b, (float*)0, 0, o_gspec, gspec_bf, DPROJ, 0, 0, BB, DPROJ, OUTD, 1, 1, 0);
  // q/k/v projections
  gemm(stream, tspec_bf, DPROJ, 0, 0, wb_wq, DPROJ, 0, 0, 1, wq_b, (float*)0, 0,
       (float*)0, tq_bf, DPROJ, 0, 0, BB, DPROJ, DPROJ, 1, 1, 0);
  gemm(stream, tspec_bf, DPROJ, 0, 0, wb_wk, DPROJ, 0, 0, 1, wk_b, (float*)0, 0,
       (float*)0, tk_bf, DPROJ, 0, 0, BB, DPROJ, DPROJ, 1, 1, 0);
  gemm(stream, tspec_bf, DPROJ, 0, 0, wb_wv, DPROJ, 0, 0, 1, wv_b, (float*)0, 0,
       (float*)0, tv_bf, DPROJ, 0, 0, BB, DPROJ, DPROJ, 1, 1, 0);
  gemm(stream, gspec_bf, DPROJ, 0, 0, wb_wq, DPROJ, 0, 0, 1, wq_b, (float*)0, 0,
       (float*)0, gq_bf, DPROJ, 0, 0, BB, DPROJ, DPROJ, 1, 1, 0);
  gemm(stream, gspec_bf, DPROJ, 0, 0, wb_wk, DPROJ, 0, 0, 1, wk_b, (float*)0, 0,
       (float*)0, gk_bf, DPROJ, 0, 0, BB, DPROJ, DPROJ, 1, 1, 0);
  gemm(stream, gspec_bf, DPROJ, 0, 0, wb_wv, DPROJ, 0, 0, 1, wv_b, (float*)0, 0,
       (float*)0, gv_bf, DPROJ, 0, 0, BB, DPROJ, DPROJ, 1, 1, 0);
  const float xscale = 1.0f / sqrtf((float)DPROJ);
  // t_cross = softmax(tq @ gk^T) @ gv
  gemm(stream, tq_bf, DPROJ, 0, 0, gk_bf, DPROJ, 0, 0, 1, (float*)0, (float*)0, 0,
       s_f, (unsigned short*)0, BB, 0, 0, BB, BB, DPROJ, 1, 1, 0);
  softmax_k<<<BB, 128, 0, stream>>>(s_f, sp_bf, BB, BB, xscale, 0, 1);
  gemm(stream, sp_bf, BB, 0, 0, gv_bf, DPROJ, 0, 0, 0, (float*)0, (float*)0, 0,
       o_tcross, tcross_bf, DPROJ, 0, 0, BB, DPROJ, BB, 1, 1, 0);
  // g_cross = softmax(gq @ tk^T) @ tv
  gemm(stream, gq_bf, DPROJ, 0, 0, tk_bf, DPROJ, 0, 0, 1, (float*)0, (float*)0, 0,
       s_f, (unsigned short*)0, BB, 0, 0, BB, BB, DPROJ, 1, 1, 0);
  softmax_k<<<BB, 128, 0, stream>>>(s_f, sp_bf, BB, BB, xscale, 0, 1);
  gemm(stream, sp_bf, BB, 0, 0, tv_bf, DPROJ, 0, 0, 0, (float*)0, (float*)0, 0,
       o_gcross, gcross_bf, DPROJ, 0, 0, BB, DPROJ, BB, 1, 1, 0);

  // ================= soft vector quantization =================
  float* qo[4]  = { out + 4 * BB * DPROJ, out + 5 * BB * DPROJ,
                    out + 6 * BB * DPROJ, out + 7 * BB * DPROJ };
  float* io[4]  = { out + 8 * BB * DPROJ,              out + 8 * BB * DPROJ + BB * KTOP,
                    out + 8 * BB * DPROJ + 2 * BB * KTOP, out + 8 * BB * DPROJ + 3 * BB * KTOP };
  const unsigned short* ebf[4] = { tspec_bf, gspec_bf, tcross_bf, gcross_bf };
  const float* efl[4] = { o_tspec, o_gspec, o_tcross, o_gcross };
  for (int v = 0; v < 4; ++v) {
    bool isText = (v == 0) || (v == 2);
    const unsigned short* cb = isText ? cbt_bf : cbg_bf;
    const float* cn = isText ? cn_t : cn_g;
    const float* cbA = isText ? cb_text : cb_graph;
    gemm(stream, ebf[v], DPROJ, 0, 0, cb, DPROJ, 0, 0, 1, (float*)0, (float*)0, 0,
         G_f, (unsigned short*)0, NCB, 0, 0, BB, NCB, DPROJ, 1, 1, 0);
    vq_k<<<1, 64, 0, stream>>>(G_f, NCB, efl[v], cn, cbA, cb_shared, SPLITR,
                               qo[v], io[v], BB, DPROJ, NCB);
  }
}
